// BiLSTM_23639499997827
// MI455X (gfx1250) — compile-verified, matrix operand order checked by reading, not run on hardware
//
#include <hip/hip_runtime.h>
#include <hip/hip_bf16.h>

// BiLSTM single step, fused GEMM(4-gate) + activations, bf16 WMMA path.
// BATCH=4096, IN=1024, HID=1024, K = IN+HID = 2048.

#define BATCH 4096
#define INDIM 1024
#define HID   1024
#define KDIM  (INDIM + HID)   // 2048

#define BM  128   // batch rows per workgroup
#define BH  32    // hidden cols per workgroup (per gate)
#define BK  64    // K elements per pipeline stage
#define LDT 72    // LDS row stride (bf16 elems): 64 + 8 pad, keeps 16B alignment

typedef __attribute__((ext_vector_type(16))) __bf16 v16bf;
typedef __attribute__((ext_vector_type(8)))  float  v8f;

// Two fp32 -> packed bf16x2 in 3 VALU: two rounding adds + one v_perm_b32.
__device__ __forceinline__ unsigned int pack2_bf16(float a, float b) {
  unsigned int ua = __float_as_uint(a) + 0x8000u;   // round-to-nearest on drop
  unsigned int ub = __float_as_uint(b) + 0x8000u;
  // bytes: [ub.3 ub.2 ua.3 ua.2] -> low16 = bf16(a), high16 = bf16(b)
  return __builtin_amdgcn_perm(ub, ua, 0x07060302u);
}

__device__ __forceinline__ float fsigmoid(float x) {
  return 1.0f / (1.0f + __expf(-x));
}

__global__ __launch_bounds__(256, 1)
void bilstm_step_kernel(const float* __restrict__ X,      // [BATCH, INDIM]
                        const float* __restrict__ Hprev,  // [BATCH, HID]
                        const float* __restrict__ Cprev,  // [BATCH, HID]
                        const float* __restrict__ W,      // [4, HID, KDIM]
                        const float* __restrict__ bias,   // [4, HID]
                        float* __restrict__ outH,         // [BATCH, HID]
                        float* __restrict__ outC)         // [BATCH, HID]
{
  // double-buffered bf16 staging: A tile (BM x BK), B tile (4*BH x BK)
  __shared__ __align__(16) unsigned short Abuf[2][BM * LDT];
  __shared__ __align__(16) unsigned short Bbuf[2][4 * BH * LDT];

  const int t        = threadIdx.x;       // 0..255
  const int lane     = t & 31;
  const int wave     = t >> 5;            // 0..7
  const int laneMod  = lane & 15;
  const int laneHalf = lane >> 4;
  const int mi2      = wave & 3;          // M sub-block (32 rows each)
  const int hj       = wave >> 2;         // H sub-block (16 cols each)

  const int m0 = blockIdx.x * BM;
  const int h0 = blockIdx.y * BH;

  // accumulators: [gate][m-subtile] ; each v8f is a 16x16 f32 C/D tile
  v8f acc[4][2];
  {
    v8f z = {0.f, 0.f, 0.f, 0.f, 0.f, 0.f, 0.f, 0.f};
    #pragma unroll
    for (int g = 0; g < 4; ++g) {
      acc[g][0] = z;
      acc[g][1] = z;
    }
  }

  // register staging for next global tile: 8 float4 per matrix per thread
  float4 aReg[8], bReg[8];

  auto load_global = [&](int ks) {
    const int kbase = ks * BK;
    const float* aptr = (kbase < INDIM) ? X : Hprev;
    const int koff = (kbase < INDIM) ? kbase : (kbase - INDIM);
    #pragma unroll
    for (int i = 0; i < 8; ++i) {
      const int idx = t + i * 256;       // 0..2047
      const int row = idx >> 4;          // 0..127
      const int c4  = idx & 15;          // float4 column within K tile
      aReg[i] = *(const float4*)(aptr + (size_t)(m0 + row) * 1024 + koff + c4 * 4);
      const int g  = row >> 5;           // gate
      const int hh = h0 + (row & 31);    // hidden col
      bReg[i] = *(const float4*)(W + ((size_t)(g * HID + hh)) * KDIM + kbase + c4 * 4);
    }
  };

  auto store_lds = [&](int buf) {
    #pragma unroll
    for (int i = 0; i < 8; ++i) {
      const int idx = t + i * 256;
      const int row = idx >> 4;
      const int c4  = idx & 15;
      uint2 pa = make_uint2(pack2_bf16(aReg[i].x, aReg[i].y),
                            pack2_bf16(aReg[i].z, aReg[i].w));
      *(uint2*)&Abuf[buf][row * LDT + c4 * 4] = pa;
      uint2 pb = make_uint2(pack2_bf16(bReg[i].x, bReg[i].y),
                            pack2_bf16(bReg[i].z, bReg[i].w));
      *(uint2*)&Bbuf[buf][row * LDT + c4 * 4] = pb;
    }
  };

  // A fragment: 16x32 bf16. lane(0..15)=row M, lanes 16..31 mirror with K+8/+24.
  auto fragA = [&](int buf, int mi, int kk) -> v16bf {
    const int row = mi2 * 32 + mi * 16 + laneMod;
    const unsigned short* base = &Abuf[buf][row * LDT + kk + laneHalf * 8];
    union { v16bf v; uint4 q[2]; } u;
    u.q[0] = *(const uint4*)(base);
    u.q[1] = *(const uint4*)(base + 16);
    return u.v;
  };
  // B fragment: 32x16 bf16, stored N-major (rows of W2) -> mirrored pattern.
  auto fragB = [&](int buf, int g, int kk) -> v16bf {
    const int row = g * 32 + hj * 16 + laneMod;
    const unsigned short* base = &Bbuf[buf][row * LDT + kk + laneHalf * 8];
    union { v16bf v; uint4 q[2]; } u;
    u.q[0] = *(const uint4*)(base);
    u.q[1] = *(const uint4*)(base + 16);
    return u.v;
  };

  const int NK = KDIM / BK;   // 32 stages

  load_global(0);
  store_lds(0);
  __syncthreads();

  for (int ks = 0; ks < NK; ++ks) {
    const int cur = ks & 1;
    const int nxt = cur ^ 1;

    if (ks + 1 < NK) {
      load_global(ks + 1);                       // hide global latency behind math
      if (ks + 2 < NK)                            // warm L2 for the weight stream
        __builtin_prefetch(W + (size_t)h0 * KDIM + (ks + 2) * BK, 0, 1);
    }

    #pragma unroll
    for (int kk = 0; kk < BK; kk += 32) {
      v16bf a0 = fragA(cur, 0, kk);
      v16bf a1 = fragA(cur, 1, kk);
      #pragma unroll
      for (int g = 0; g < 4; ++g) {
        v16bf bmat = fragB(cur, g, kk);
        acc[g][0] = __builtin_amdgcn_wmma_f32_16x16x32_bf16(
            false, a0, false, bmat, (short)0, acc[g][0], false, false);
        acc[g][1] = __builtin_amdgcn_wmma_f32_16x16x32_bf16(
            false, a1, false, bmat, (short)0, acc[g][1], false, false);
      }
    }

    if (ks + 1 < NK) store_lds(nxt);
    __syncthreads();
  }

  // ---- fused LSTM epilogue: all 4 gates for (m, h) live in this lane ----
  const int hcol = h0 + hj * 16 + laneMod;
  float bv[4];
  #pragma unroll
  for (int g = 0; g < 4; ++g) bv[g] = bias[g * HID + hcol];

  #pragma unroll
  for (int mi = 0; mi < 2; ++mi) {
    #pragma unroll
    for (int r = 0; r < 8; ++r) {
      const int m = m0 + mi2 * 32 + mi * 16 + r + laneHalf * 8;
      const float gf = fsigmoid(acc[0][mi][r] + bv[0]);
      const float gi = fsigmoid(acc[1][mi][r] + bv[1]);
      const float go = fsigmoid(acc[2][mi][r] + bv[2]);
      const float gc = tanhf(acc[3][mi][r] + bv[3]);
      const float c_old = Cprev[(size_t)m * HID + hcol];
      const float cn = gf * c_old + gi * gc;
      const float hn = go * tanhf(cn);
      outH[(size_t)m * HID + hcol] = hn;
      outC[(size_t)m * HID + hcol] = cn;
    }
  }
}

extern "C" void kernel_launch(void* const* d_in, const int* in_sizes, int n_in,
                              void* d_out, int out_size, void* d_ws, size_t ws_size,
                              hipStream_t stream) {
  const float* input_f = (const float*)d_in[0];
  const float* input_b = (const float*)d_in[1];
  const float* Hf      = (const float*)d_in[2];
  const float* Cf      = (const float*)d_in[3];
  const float* Hb      = (const float*)d_in[4];
  const float* Cb      = (const float*)d_in[5];
  const float* Wf      = (const float*)d_in[6];
  const float* bf_     = (const float*)d_in[7];
  const float* Wb      = (const float*)d_in[8];
  const float* bb_     = (const float*)d_in[9];
  float* out = (float*)d_out;
  const size_t N = (size_t)BATCH * HID;

  dim3 grid(BATCH / BM, HID / BH, 1);
  dim3 block(256);
  // outputs in tuple order: h_f, c_f, h_b, c_b
  bilstm_step_kernel<<<grid, block, 0, stream>>>(input_f, Hf, Cf, Wf, bf_,
                                                 out + 0 * N, out + 1 * N);
  bilstm_step_kernel<<<grid, block, 0, stream>>>(input_b, Hb, Cb, Wb, bb_,
                                                 out + 2 * N, out + 3 * N);
}
